// GPT2SelfAttention_30485677867662
// MI455X (gfx1250) — matrix operand order repaired and checked by reading, not compiled, not run on hardware
//
#include <hip/hip_runtime.h>

// ---------------------------------------------------------------------------
// CDNA5 / gfx1250 GPT-2 self-attention.
// All matrix math: V_WMMA_F32_16X16X32_BF16 (fp32 accumulate).
// Dataflow: one-time fp32->bf16 conversion pre-pass (weights transposed),
// then all GEMM/attention tiles are raw bf16 moved with async global->LDS
// copies (double buffered, ASYNCcnt-synchronized).
// ---------------------------------------------------------------------------

typedef __attribute__((ext_vector_type(16))) __bf16 bf16x16;
typedef __attribute__((ext_vector_type(8)))  __bf16 bf16x8;
typedef __attribute__((ext_vector_type(8)))  float  f32x8;
typedef __attribute__((ext_vector_type(4)))  float  f32x4;
typedef __attribute__((ext_vector_type(4)))  int    i32x4;

union Frag {
    bf16x16 v;
    bf16x8  h[2];
};

__device__ __forceinline__ f32x8 wmma_bf16(const Frag& a, const Frag& b, f32x8 c) {
    return __builtin_amdgcn_wmma_f32_16x16x32_bf16(false, a.v, false, b.v,
                                                   (short)0, c, false, false);
}

// ------------------------- async global->LDS copy --------------------------
#if __has_builtin(__builtin_amdgcn_global_load_async_to_lds_b128)
#define ASYNC_LDS 1
#else
#define ASYNC_LDS 0
#endif

__device__ __forceinline__ void copy_b128(__bf16* lds, const __bf16* g) {
#if ASYNC_LDS
    typedef __attribute__((address_space(1))) i32x4 gv_t;   // global, 64-bit ptr
    typedef __attribute__((address_space(3))) i32x4 lv_t;   // LDS, 32-bit ptr
    __builtin_amdgcn_global_load_async_to_lds_b128(
        (gv_t*)(uintptr_t)g,
        (lv_t*)(uint32_t)(uintptr_t)lds,   // flat LDS addr[31:0] == LDS offset
        0, 0);
#else
    *(bf16x8*)lds = *(const bf16x8*)g;
#endif
}

template<int N>
__device__ __forceinline__ void wait_async() {
#if ASYNC_LDS
#if __has_builtin(__builtin_amdgcn_s_wait_asynccnt)
    __builtin_amdgcn_s_wait_asynccnt(N);
#else
    asm volatile("s_wait_asynccnt %0" :: "i"(N) : "memory");
#endif
#endif
}

// ---------------------------------------------------------------------------
// Pre-pass 1: fp32 -> bf16 elementwise (hidden states).
// ---------------------------------------------------------------------------
__global__ __launch_bounds__(256)
void cvt_bf16_kernel(const float* __restrict__ src, __bf16* __restrict__ dst, int n)
{
    int i = (blockIdx.x * 256 + threadIdx.x) * 4;
    if (i < n) {
        f32x4 v = *(const f32x4*)(src + i);
        dst[i + 0] = (__bf16)v[0];
        dst[i + 1] = (__bf16)v[1];
        dst[i + 2] = (__bf16)v[2];
        dst[i + 3] = (__bf16)v[3];
    }
}

// ---------------------------------------------------------------------------
// Pre-pass 2: W[K][N] fp32 -> WT[N][K] bf16 (32x32 LDS tile transpose).
// ---------------------------------------------------------------------------
__global__ __launch_bounds__(256)
void cvt_transpose_kernel(const float* __restrict__ W, __bf16* __restrict__ WT,
                          int K, int N)
{
    __shared__ float t[32][33];
    const int nb = blockIdx.x * 32;
    const int kb = blockIdx.y * 32;
    const int tx = threadIdx.x & 31;
    const int ty = threadIdx.x >> 5;    // 0..7
    #pragma unroll
    for (int r = 0; r < 32; r += 8)
        t[ty + r][tx] = W[(size_t)(kb + ty + r) * N + nb + tx];
    __syncthreads();
    #pragma unroll
    for (int r = 0; r < 32; r += 8)
        WT[(size_t)(nb + ty + r) * K + kb + tx] = (__bf16)t[tx][ty + r];
}

// ---------------------------------------------------------------------------
// GEMM: C[M,N] = A[M,K](bf16) @ BT[N,K](bf16, pre-transposed) + bias.
// Block tile 64x128, 8 waves (2x4), wave tile 32x32 (2x2 WMMA), BK=32.
// Double-buffered LDS fed by async b128 copies (3 per wave per stage).
// MODE 0: bf16 scatter into qkv ws: q,k = [B,H,S,hd]; v = [B,H,hd,S]
// MODE 1: fp32 row-major store
// ---------------------------------------------------------------------------
template<int MODE>
__global__ __launch_bounds__(256)
void gemm_bf16_wmma(const __bf16* __restrict__ A, const __bf16* __restrict__ BT,
                    const float* __restrict__ bias, void* __restrict__ Cout,
                    int M, int N, int K)
{
    __shared__ __align__(16) __bf16 As[2][64][40];
    __shared__ __align__(16) __bf16 Bs[2][128][40];

    const int tid  = threadIdx.x;
    const int lane = tid & 31;
    const int wid  = tid >> 5;
    const int wm   = wid >> 2;
    const int wn   = wid & 3;
    const int hm   = lane & 15;
    const int hs   = lane >> 4;

    const int blockM = blockIdx.y * 64;
    const int blockN = blockIdx.x * 128;

    // per-wave async stage: A 64x32 (1 instr), B 128x32 (2 instr)
    auto stage = [&](int buf, int k0) {
        {
            int row = wid * 8 + (lane >> 2);
            int ch  = (lane & 3) * 8;
            copy_b128(&As[buf][row][ch], A + (size_t)(blockM + row) * K + k0 + ch);
        }
        #pragma unroll
        for (int j = 0; j < 2; ++j) {
            int row = wid * 16 + j * 8 + (lane >> 2);
            int ch  = (lane & 3) * 8;
            copy_b128(&Bs[buf][row][ch], BT + (size_t)(blockN + row) * K + k0 + ch);
        }
    };

    f32x8 acc[2][2] = {};
    const int nk = K >> 5;

    stage(0, 0);
    for (int kk = 0; kk < nk; ++kk) {
        const int buf = kk & 1;
        if (kk + 1 < nk) { stage(buf ^ 1, (kk + 1) * 32); wait_async<3>(); }
        else             { wait_async<0>(); }
        __syncthreads();

        Frag af[2], bfr[2];
        #pragma unroll
        for (int i = 0; i < 2; ++i) {
            int r = wm * 32 + i * 16 + hm;
            af[i].h[0] = *(const bf16x8*)&As[buf][r][hs * 8];
            af[i].h[1] = *(const bf16x8*)&As[buf][r][16 + hs * 8];
        }
        #pragma unroll
        for (int j = 0; j < 2; ++j) {
            int c = wn * 32 + j * 16 + hm;
            bfr[j].h[0] = *(const bf16x8*)&Bs[buf][c][hs * 8];
            bfr[j].h[1] = *(const bf16x8*)&Bs[buf][c][16 + hs * 8];
        }
        #pragma unroll
        for (int i = 0; i < 2; ++i)
            #pragma unroll
            for (int j = 0; j < 2; ++j)
                acc[i][j] = wmma_bf16(af[i], bfr[j], acc[i][j]);
        __syncthreads();
    }

    #pragma unroll
    for (int i = 0; i < 2; ++i) {
        #pragma unroll
        for (int j = 0; j < 2; ++j) {
            int col = blockN + wn * 32 + j * 16 + hm;
            float bv = bias[col];
            #pragma unroll
            for (int r = 0; r < 8; ++r) {
                int row = blockM + wm * 32 + i * 16 + r + 8 * hs;
                float val = acc[i][j][r] + bv;
                if (MODE == 0) {
                    __bf16* C = (__bf16*)Cout;
                    int part = col >> 10, dp = col & 1023;
                    int h = dp >> 6, dd = dp & 63;
                    int bb = row >> 10, ss = row & 1023;
                    size_t plane = (size_t)part * 64 * 1024 * 64;
                    size_t dst;
                    if (part == 2)  // V transposed: [b,h,hd,S]
                        dst = plane + (((size_t)bb * 16 + h) * 64 + dd) * 1024 + ss;
                    else            // Q,K: [b,h,S,hd]
                        dst = plane + (((size_t)bb * 16 + h) * 1024 + ss) * 64 + dd;
                    C[dst] = (__bf16)val;
                } else {
                    ((float*)Cout)[(size_t)row * N + col] = val;
                }
            }
        }
    }
}

// ---------------------------------------------------------------------------
// Flash attention: grid (B*H=64, S/64=16), 128 threads = 4 waves.
// All-bf16 inputs; K and pre-transposed V double-buffered via async copies.
// Output bf16 [B,S,D] with heads merged.
// ---------------------------------------------------------------------------
__global__ __launch_bounds__(128)
void attn_flash_wmma(const __bf16* __restrict__ qkv, __bf16* __restrict__ aout)
{
    constexpr int S = 1024, HD = 64;
    const int bh = blockIdx.x;
    const int qb = blockIdx.y;

    const __bf16* Qg = qkv + (size_t)bh * S * HD;
    const __bf16* Kg = qkv + (size_t)64 * S * HD + (size_t)bh * S * HD;
    const __bf16* Vt = qkv + (size_t)2 * 64 * S * HD + (size_t)bh * HD * S;

    __shared__ __align__(16) __bf16 Qs[64][72];
    __shared__ __align__(16) __bf16 Ks[2][32][72];
    __shared__ __align__(16) __bf16 VT[2][64][40];
    __shared__ __align__(16) __bf16 Ps[4][16][40];

    const int tid  = threadIdx.x;
    const int lane = tid & 31;
    const int w    = tid >> 5;
    const int hm   = lane & 15;
    const int hs   = lane >> 4;

    // stage Q tile 64x64 (synchronous, once)
    #pragma unroll
    for (int it = 0; it < 4; ++it) {
        int idx = tid + it * 128;        // 0..511 chunk slots
        int row = idx >> 3;
        int ch  = (idx & 7) * 8;
        *(bf16x8*)&Qs[row][ch] =
            *(const bf16x8*)(Qg + (size_t)(qb * 64 + row) * HD + ch);
    }

    // per-wave async stage of K (2 instr) + VT (2 instr)
    auto stage_kv = [&](int buf, int kb) {
        #pragma unroll
        for (int j = 0; j < 2; ++j) {
            int row = (w * 2 + j) * 4 + (lane >> 3);   // key 0..31
            int ch  = (lane & 7) * 8;
            copy_b128(&Ks[buf][row][ch], Kg + (size_t)(kb * 32 + row) * HD + ch);
        }
        #pragma unroll
        for (int j = 0; j < 2; ++j) {
            int row = (w * 2 + j) * 8 + (lane >> 2);   // hd 0..63
            int ch  = (lane & 3) * 8;
            copy_b128(&VT[buf][row][ch], Vt + (size_t)row * S + kb * 32 + ch);
        }
    };

    __syncthreads();   // Qs visible

    Frag qf[2];
    #pragma unroll
    for (int c = 0; c < 2; ++c) {
        qf[c].h[0] = *(const bf16x8*)&Qs[w * 16 + hm][c * 32 + hs * 8];
        qf[c].h[1] = *(const bf16x8*)&Qs[w * 16 + hm][c * 32 + 16 + hs * 8];
    }

    f32x8 o[4] = {};
    float mrow[8], lrow[8];
    #pragma unroll
    for (int r = 0; r < 8; ++r) { mrow[r] = -1e30f; lrow[r] = 0.0f; }

    const int qrow0 = qb * 64 + w * 16;
    const int nkb   = 2 * qb + 2;

    stage_kv(0, 0);
    for (int kb = 0; kb < nkb; ++kb) {
        const int buf = kb & 1;
        if (kb + 1 < nkb) { stage_kv(buf ^ 1, kb + 1); wait_async<4>(); }
        else              { wait_async<0>(); }
        __syncthreads();

        if (kb * 32 <= qrow0 + 15) {          // wave-uniform
            f32x8 sc[2];
            #pragma unroll
            for (int sub = 0; sub < 2; ++sub) {
                f32x8 s = {};
                Frag kf;
                kf.h[0] = *(const bf16x8*)&Ks[buf][sub * 16 + hm][hs * 8];
                kf.h[1] = *(const bf16x8*)&Ks[buf][sub * 16 + hm][16 + hs * 8];
                s = wmma_bf16(qf[0], kf, s);
                kf.h[0] = *(const bf16x8*)&Ks[buf][sub * 16 + hm][32 + hs * 8];
                kf.h[1] = *(const bf16x8*)&Ks[buf][sub * 16 + hm][32 + 16 + hs * 8];
                s = wmma_bf16(qf[1], kf, s);
                int key = kb * 32 + sub * 16 + hm;
                #pragma unroll
                for (int r = 0; r < 8; ++r) {
                    int qr = qrow0 + r + 8 * hs;
                    float v = s[r] * 0.125f;
                    sc[sub][r] = (key <= qr) ? v : -10000.0f;
                }
            }
            #pragma unroll
            for (int r = 0; r < 8; ++r) {
                float mx = fmaxf(sc[0][r], sc[1][r]);
                #pragma unroll
                for (int off = 1; off < 16; off <<= 1)
                    mx = fmaxf(mx, __shfl_xor(mx, off, 16));
                float mn   = fmaxf(mrow[r], mx);
                float corr = __expf(mrow[r] - mn);
                float p0   = __expf(sc[0][r] - mn);
                float p1   = __expf(sc[1][r] - mn);
                float ps   = p0 + p1;
                #pragma unroll
                for (int off = 1; off < 16; off <<= 1)
                    ps += __shfl_xor(ps, off, 16);
                lrow[r] = lrow[r] * corr + ps;
                mrow[r] = mn;
                #pragma unroll
                for (int t = 0; t < 4; ++t) o[t][r] *= corr;
                Ps[w][r + 8 * hs][hm]      = (__bf16)p0;
                Ps[w][r + 8 * hs][16 + hm] = (__bf16)p1;
            }
            asm volatile("" ::: "memory");   // same-wave DS ops are in-order
            Frag pf;
            pf.h[0] = *(const bf16x8*)&Ps[w][hm][hs * 8];
            pf.h[1] = *(const bf16x8*)&Ps[w][hm][16 + hs * 8];
            #pragma unroll
            for (int t = 0; t < 4; ++t) {
                Frag vf;
                vf.h[0] = *(const bf16x8*)&VT[buf][t * 16 + hm][hs * 8];
                vf.h[1] = *(const bf16x8*)&VT[buf][t * 16 + hm][16 + hs * 8];
                o[t] = wmma_bf16(pf, vf, o[t]);
            }
        }
        __syncthreads();
    }

    const int bb = bh >> 4, hh = bh & 15;
    #pragma unroll
    for (int r = 0; r < 8; ++r) {
        float inv = 1.0f / lrow[r];
        int srow  = qrow0 + r + 8 * hs;
        #pragma unroll
        for (int t = 0; t < 4; ++t) {
            aout[((size_t)bb * 1024 + srow) * 1024 + hh * 64 + t * 16 + hm] =
                (__bf16)(o[t][r] * inv);
        }
    }
}

// ---------------------------------------------------------------------------
// Launch. ws layout (bf16 elements, all 16B aligned):
//   hbf  [4096*1024]      hidden bf16
//   wt1  [3072*1024]      c_attn_w^T bf16
//   wt2  [1024*1024]      c_proj_w^T bf16
//   qkvb [3*64*1024*64]   q,k:[B,H,S,hd]  v:[B,H,hd,S]
//   aob  [4096*1024]      attention out bf16 [B,S,D]
// total ~48 MB (fits comfortably in MI455X's 192 MB L2)
// ---------------------------------------------------------------------------
extern "C" void kernel_launch(void* const* d_in, const int* in_sizes, int n_in,
                              void* d_out, int out_size, void* d_ws, size_t ws_size,
                              hipStream_t stream)
{
    (void)in_sizes; (void)n_in; (void)out_size; (void)ws_size;
    const float* hidden   = (const float*)d_in[0];
    const float* c_attn_w = (const float*)d_in[1];
    const float* c_attn_b = (const float*)d_in[2];
    const float* c_proj_w = (const float*)d_in[3];
    const float* c_proj_b = (const float*)d_in[4];
    float* out = (float*)d_out;

    __bf16* hbf  = (__bf16*)d_ws;
    __bf16* wt1  = hbf  + (size_t)4096 * 1024;
    __bf16* wt2  = wt1  + (size_t)3072 * 1024;
    __bf16* qkvb = wt2  + (size_t)1024 * 1024;
    __bf16* aob  = qkvb + (size_t)3 * 64 * 1024 * 64;

    // pre-pass conversions
    hipLaunchKernelGGL(cvt_bf16_kernel, dim3(4096), dim3(256), 0, stream,
                       hidden, hbf, 4096 * 1024);
    hipLaunchKernelGGL(cvt_transpose_kernel, dim3(96, 32), dim3(256), 0, stream,
                       c_attn_w, wt1, 1024, 3072);
    hipLaunchKernelGGL(cvt_transpose_kernel, dim3(32, 32), dim3(256), 0, stream,
                       c_proj_w, wt2, 1024, 1024);

    // QKV projection -> bf16 scatter (V pre-transposed)
    hipLaunchKernelGGL((gemm_bf16_wmma<0>), dim3(24, 64), dim3(256), 0, stream,
                       hbf, wt1, c_attn_b, (void*)qkvb, 4096, 3072, 1024);

    // flash attention -> bf16 [B,S,D]
    hipLaunchKernelGGL(attn_flash_wmma, dim3(64, 16), dim3(128), 0, stream,
                       qkvb, aob);

    // output projection -> fp32 d_out
    hipLaunchKernelGGL((gemm_bf16_wmma<1>), dim3(8, 64), dim3(256), 0, stream,
                       aob, wt2, c_proj_b, (void*)out, 4096, 1024, 1024);
}
